// Attention_49993419325755
// MI455X (gfx1250) — compile-verified, matrix operand order checked by reading, not run on hardware
//
#include <hip/hip_runtime.h>
#include <hip/hip_bf16.h>

// ---------------------------------------------------------------------------
// MI455X (gfx1250) implementation.
// Dominant cost: 16 LSTMs x 192 steps x GEMM(10400x512x128) ~ 4.2 TFLOP.
// K1 computes z^T = Whh x h^T per step:
//   A (Whh) fragments live in VGPRs for all 192 steps (zero LDS traffic),
//   B (h^T) fragments are A-pattern loads from row-major h in LDS,
//   h double-buffered in LDS, one barrier per step.
// Attention matmuls (scores, attn@v) also routed through WMMA.
// ---------------------------------------------------------------------------

typedef __attribute__((ext_vector_type(16))) __bf16 v16bf;
typedef __attribute__((ext_vector_type(8)))  __bf16 v8bf;
typedef __attribute__((ext_vector_type(8)))  float  v8f;

#define NSEQ   10400       // B*S
#define BATCH  32
#define SEQ    325
#define TS     192         // L (timesteps / value dim)
#define DM     128         // d_model
#define G4     512         // 4*d_model
#define SP     352         // padded seq (11*32)
#define QKROWS 10432       // padded rows for q/k buffers
#define HSTR   136         // padded row stride (ushorts) for h tiles
#define NEGBIG -1000000000.0f

__device__ __forceinline__ unsigned short f2bf(float f) {
  union { float f; unsigned int u; } v; v.f = f;
  unsigned int r = v.u + 0x7FFFu + ((v.u >> 16) & 1u);   // RNE
  return (unsigned short)(r >> 16);
}
__device__ __forceinline__ __bf16 bfc(float f) {
  unsigned short u = f2bf(f);
  return __builtin_bit_cast(__bf16, u);
}
__device__ __forceinline__ float sigm(float z) { return 1.f / (1.f + __expf(-z)); }

// A/B fragment with the 16x16x32 bf16 lane layout: 8 halves at p, 8 at p+16
__device__ __forceinline__ v16bf ld_frag16(const unsigned short* p) {
  v8bf lo = *(const v8bf*)p;
  v8bf hi = *(const v8bf*)(p + 16);
  return __builtin_shufflevector(lo, hi, 0,1,2,3,4,5,6,7,8,9,10,11,12,13,14,15);
}

// ---------------------------------------------------------------------------
// K1: q/k LSTMs. grid = (82 seq-tiles of 128, 16 lstm ids). block = 256.
// Wave w owns hidden block j=w (cols j*16..j*16+15 of h) and the 4 gate
// row-blocks {j, 8+j, 16+j, 24+j} of Whh, held as A-fragments in registers.
// LDS: h double buffer 68KB | Xc 16.5KB  (~85KB)
// ---------------------------------------------------------------------------
__global__ __launch_bounds__(256)
void lstm_qk(const float* __restrict__ x,
             const float* __restrict__ qWih, const float* __restrict__ qWhh,
             const float* __restrict__ qbih, const float* __restrict__ qbhh,
             const float* __restrict__ kWih, const float* __restrict__ kWhh,
             const float* __restrict__ kbih, const float* __restrict__ kbhh,
             unsigned short* __restrict__ qk)
{
  extern __shared__ char smem[];
  unsigned short* Hb = (unsigned short*)smem;              // [2][128][HSTR]
  float*          Xc = (float*)(smem + 2 * 128 * HSTR * 2); // [128][33]

  const int tid  = threadIdx.x;
  const int wave = tid >> 5;
  const int lane = tid & 31;
  const int type = blockIdx.y >> 3;
  const int head = blockIdx.y & 7;
  const long base_seq = (long)blockIdx.x * 128;

  const float* Whh = (type == 0) ? qWhh : kWhh;
  const float* Wih = (type == 0) ? qWih : kWih;
  const float* bih = (type == 0) ? qbih : kbih;
  const float* bhh = (type == 0) ? qbhh : kbhh;
  const float* WhhH = Whh + (size_t)head * G4 * DM;

  const int j      = wave;                 // hidden 16-block owned by wave
  const int lane15 = lane & 15;
  const int half8  = (lane >= 16) ? 8 : 0;

  // ---- loop-invariant A-fragments of Whh (f32 -> bf16 in registers) ----
  v16bf wfrag[4][4];                       // [gate][kb]
#pragma unroll
  for (int g = 0; g < 4; ++g) {
#pragma unroll
    for (int kb = 0; kb < 4; ++kb) {
      const float* wp = WhhH + (size_t)((g * 8 + j) * 16 + lane15) * DM
                             + kb * 32 + half8;
      v16bf a;
#pragma unroll
      for (int i = 0; i < 8; ++i) a[i] = bfc(wp[i]);
#pragma unroll
      for (int i = 0; i < 8; ++i) a[8 + i] = bfc(wp[16 + i]);
      wfrag[g][kb] = a;
    }
  }
  // ---- hoisted per-thread Wih / bias: col = g*128 + j*16 + r + half8 ----
  float wiR[4][8], bbR[4][8];
#pragma unroll
  for (int g = 0; g < 4; ++g)
#pragma unroll
    for (int r = 0; r < 8; ++r) {
      int n = g * 128 + j * 16 + r + half8;
      wiR[g][r] = Wih[head * G4 + n];
      bbR[g][r] = bih[head * G4 + n] + bhh[head * G4 + n];
    }

  for (int idx = tid; idx < 128 * HSTR; idx += 256) Hb[idx] = 0;  // zero buf0

  float creg[8][8];
#pragma unroll
  for (int mb = 0; mb < 8; ++mb)
#pragma unroll
    for (int r = 0; r < 8; ++r) creg[mb][r] = 0.f;

  for (int t = 0; t < TS; ++t) {
    __syncthreads();                       // step-t reads see step-(t-1) writes
    if ((t & 31) == 0) {                   // refill 32-step x chunk
      for (int idx = tid; idx < 128 * 32; idx += 256) {
        int row = idx >> 5, col = idx & 31;
        long seq = base_seq + row;
        Xc[row * 33 + col] = (seq < NSEQ) ? x[seq * TS + t + col] : 0.f;
      }
      __syncthreads();
    }
    const unsigned short* Hr = Hb + (t & 1) * 128 * HSTR;
    unsigned short*       Hw = Hb + ((t + 1) & 1) * 128 * HSTR;

#pragma unroll
    for (int mb = 0; mb < 8; ++mb) {       // 16 sequence rows per block
      v16bf hfrag[4];
#pragma unroll
      for (int kb = 0; kb < 4; ++kb)
        hfrag[kb] = ld_frag16(Hr + (mb * 16 + lane15) * HSTR
                                 + kb * 32 + half8);
      v8f zi = {0,0,0,0,0,0,0,0}, zf = zi, zg = zi, zo = zi;
#pragma unroll
      for (int kb = 0; kb < 4; ++kb) {
        zi = __builtin_amdgcn_wmma_f32_16x16x32_bf16(false, wfrag[0][kb], false, hfrag[kb], (short)0, zi, false, false);
        zf = __builtin_amdgcn_wmma_f32_16x16x32_bf16(false, wfrag[1][kb], false, hfrag[kb], (short)0, zf, false, false);
        zg = __builtin_amdgcn_wmma_f32_16x16x32_bf16(false, wfrag[2][kb], false, hfrag[kb], (short)0, zg, false, false);
        zo = __builtin_amdgcn_wmma_f32_16x16x32_bf16(false, wfrag[3][kb], false, hfrag[kb], (short)0, zo, false, false);
      }
      // D tile is [n][m]: lane -> m = mb*16+lane15 ; r -> n_local = r+half8
      const float xm = Xc[(mb * 16 + lane15) * 33 + (t & 31)];
#pragma unroll
      for (int r = 0; r < 8; ++r) {
        float vi = zi[r] + xm * wiR[0][r] + bbR[0][r];
        float vf = zf[r] + xm * wiR[1][r] + bbR[1][r];
        float vg = zg[r] + xm * wiR[2][r] + bbR[2][r];
        float vo = zo[r] + xm * wiR[3][r] + bbR[3][r];
        float c  = sigm(vf) * creg[mb][r] + sigm(vi) * tanhf(vg);
        creg[mb][r] = c;
        float h  = sigm(vo) * tanhf(c);
        Hw[(mb * 16 + lane15) * HSTR + j * 16 + r + half8] = f2bf(h);
      }
    }
  }
  __syncthreads();

  // final h lives in buffer (TS & 1) == 0
  unsigned short* dst = qk + ((size_t)(type * 8 + head) * QKROWS + base_seq) * DM;
  for (int idx = tid; idx < 128 * DM; idx += 256) {
    int row = idx >> 7, col = idx & 127;
    if (base_seq + row < NSEQ) dst[row * DM + col] = Hb[row * HSTR + col];
  }
}

// ---------------------------------------------------------------------------
// K2: hidden-size-1 v LSTM, one thread per (head, seq). Writes vT (bf16)
// as [H][B][192][SP] so K4 B-fragments are contiguous row loads.
// ---------------------------------------------------------------------------
__global__ __launch_bounds__(256)
void lstm_v(const float* __restrict__ x,
            const float* __restrict__ vWih, const float* __restrict__ vWhh,
            const float* __restrict__ vbih, const float* __restrict__ vbhh,
            unsigned short* __restrict__ vT)
{
  int gid = blockIdx.x * blockDim.x + threadIdx.x;
  if (gid >= 8 * NSEQ) return;
  int head = gid / NSEQ, seq = gid % NSEQ;
  int b = seq / SEQ, s = seq % SEQ;
  float wih[4], whh[4], bb[4];
#pragma unroll
  for (int g = 0; g < 4; ++g) {
    wih[g] = vWih[head * 4 + g];
    whh[g] = vWhh[head * 4 + g];
    bb[g]  = vbih[head * 4 + g] + vbhh[head * 4 + g];
  }
  float h = 0.f, c = 0.f;
  unsigned short* dst = vT + (size_t)(head * BATCH + b) * TS * SP + s;
  for (int t = 0; t < TS; ++t) {
    float xt = x[(size_t)seq * TS + t];
    float zi = xt * wih[0] + h * whh[0] + bb[0];
    float zf = xt * wih[1] + h * whh[1] + bb[1];
    float zg = xt * wih[2] + h * whh[2] + bb[2];
    float zo = xt * wih[3] + h * whh[3] + bb[3];
    c = sigm(zf) * c + sigm(zi) * tanhf(zg);
    h = sigm(zo) * tanhf(c);
    dst[(size_t)t * SP] = f2bf(h);
  }
}

// ---------------------------------------------------------------------------
// K3: scores = q k^T / sqrt(d); leaky; +mask; softmax; -> attn (bf16, SP cols)
// grid = (6 row-tiles of 64, 256 (head,b) pairs)
// ---------------------------------------------------------------------------
__global__ __launch_bounds__(256)
void attn_scores(const unsigned short* __restrict__ qk,
                 const float* __restrict__ graph,
                 unsigned short* __restrict__ attn)
{
  extern __shared__ char smem[];
  unsigned short* Qt = (unsigned short*)smem;                 // [64][136]
  unsigned short* Kt = (unsigned short*)(smem + 17408);       // [32][136]
  float*          Sc = (float*)(smem + 17408 + 8704);         // [64][360]

  const int tid = threadIdx.x, wave = tid >> 5, lane = tid & 31;
  const int head = blockIdx.y >> 5, b = blockIdx.y & 31;
  const int srow0 = blockIdx.x * 64;
  const unsigned short* qbuf = qk + (size_t)head * QKROWS * DM;
  const unsigned short* kbuf = qk + (size_t)(8 + head) * QKROWS * DM;

  for (int idx = tid; idx < 64 * DM; idx += 256) {
    int row = idx >> 7, col = idx & 127;
    int s = srow0 + row;
    Qt[row * HSTR + col] = (s < SEQ) ? qbuf[((size_t)(b * SEQ + s)) * DM + col] : 0;
  }

  const int rb  = wave & 3;     // row block
  const int nbg = wave >> 2;    // local col block (0/1)
  const int m   = (lane >= 16) ? 8 : 0, n = lane & 15;

  for (int ct = 0; ct < 11; ++ct) {   // 32 score-cols per iter
    __syncthreads();
    for (int idx = tid; idx < 32 * DM; idx += 256) {
      int row = idx >> 7, col = idx & 127;
      int tcol = ct * 32 + row;
      Kt[row * HSTR + col] = (tcol < SEQ) ? kbuf[((size_t)(b * SEQ + tcol)) * DM + col] : 0;
    }
    __syncthreads();
    v8f acc = {0,0,0,0,0,0,0,0};
#pragma unroll
    for (int kb = 0; kb < 4; ++kb) {
      v16bf a  = ld_frag16(Qt + (rb  * 16 + (lane & 15)) * HSTR + kb * 32 + ((lane >= 16) ? 8 : 0));
      v16bf bm = ld_frag16(Kt + (nbg * 16 + (lane & 15)) * HSTR + kb * 32 + ((lane >= 16) ? 8 : 0));
      acc = __builtin_amdgcn_wmma_f32_16x16x32_bf16(false, a, false, bm, (short)0, acc, false, false);
    }
#pragma unroll
    for (int r = 0; r < 8; ++r) {
      float v = acc[r] * 0.08838834764831845f;     // 1/sqrt(128)
      v = (v >= 0.f) ? v : 0.2f * v;               // leaky
      Sc[(rb * 16 + m + r) * 360 + ct * 32 + nbg * 16 + n] = v;
    }
  }
  __syncthreads();

  if (tid < 64) {
    int s = srow0 + tid;
    if (s < SEQ) {
      float* row = Sc + tid * 360;
      float mx = -3.4e38f;
      for (int t = 0; t < SEQ; ++t) {
        float g = graph[(size_t)t * SEQ + s] + ((s == t) ? 1.f : 0.f);   // graph.T + I
        float v = row[t] + ((g == 0.f) ? NEGBIG : 0.f);
        row[t] = v;
        mx = fmaxf(mx, v);
      }
      float sum = 0.f;
      for (int t = 0; t < SEQ; ++t) { float e = __expf(row[t] - mx); row[t] = e; sum += e; }
      float inv = 1.f / sum;
      unsigned short* arow = attn + ((size_t)(head * BATCH + b) * SP + s) * SP;
      for (int t = 0; t < SEQ; ++t) arow[t] = f2bf(row[t] * inv);
      for (int t = SEQ; t < SP; ++t) arow[t] = 0;
    }
  }
}

// ---------------------------------------------------------------------------
// K4: out = leaky(attn @ v) -> d_out[B][S][L][H] (f32)
// grid = (6 row-tiles of 64, 256 (head,b) pairs)
// ---------------------------------------------------------------------------
__global__ __launch_bounds__(256)
void attn_out(const unsigned short* __restrict__ attn,
              const unsigned short* __restrict__ vT,
              float* __restrict__ out)
{
  extern __shared__ char smem[];
  unsigned short* At = (unsigned short*)smem;              // [64][360]
  unsigned short* Vc = (unsigned short*)(smem + 46080);    // [192][40]

  const int tid = threadIdx.x, wave = tid >> 5, lane = tid & 31;
  const int head = blockIdx.y >> 5, b = blockIdx.y & 31;
  const int srow0 = blockIdx.x * 64;
  const unsigned short* abuf = attn + (size_t)(head * BATCH + b) * SP * SP;
  const unsigned short* vbuf = vT   + (size_t)(head * BATCH + b) * TS * SP;

  for (int idx = tid; idx < 64 * SP; idx += 256) {
    int row = idx / SP, col = idx % SP;
    int s = srow0 + row;
    At[row * 360 + col] = (s < SP) ? abuf[(size_t)s * SP + col] : 0;
  }

  const int rb = wave & 3;
  const int ng = (wave >> 2) * 6;            // 6 output col-blocks per wave
  const int m  = (lane >= 16) ? 8 : 0, n = lane & 15;
  v8f acc[6];
#pragma unroll
  for (int q = 0; q < 6; ++q) acc[q] = (v8f){0,0,0,0,0,0,0,0};

  for (int kb = 0; kb < 11; ++kb) {          // 32 k (t) per iter
    __syncthreads();
    for (int idx = tid; idx < TS * 32; idx += 256) {
      int row = idx >> 5, col = idx & 31;
      Vc[row * 40 + col] = vbuf[(size_t)row * SP + kb * 32 + col];
    }
    __syncthreads();
    v16bf a = ld_frag16(At + (rb * 16 + (lane & 15)) * 360 + kb * 32 + ((lane >= 16) ? 8 : 0));
#pragma unroll
    for (int q = 0; q < 6; ++q) {
      v16bf bm = ld_frag16(Vc + ((ng + q) * 16 + (lane & 15)) * 40 + ((lane >= 16) ? 8 : 0));
      acc[q] = __builtin_amdgcn_wmma_f32_16x16x32_bf16(false, a, false, bm, (short)0, acc[q], false, false);
    }
  }

#pragma unroll
  for (int q = 0; q < 6; ++q) {
    int l0 = (ng + q) * 16 + n;
#pragma unroll
    for (int r = 0; r < 8; ++r) {
      int s = srow0 + rb * 16 + m + r;
      if (s < SEQ) {
        float v = acc[q][r];
        v = (v >= 0.f) ? v : 0.2f * v;       // final leaky
        out[(((size_t)b * SEQ + s) * TS + l0) * 8 + head] = v;
      }
    }
  }
}

// ---------------------------------------------------------------------------
extern "C" void kernel_launch(void* const* d_in, const int* in_sizes, int n_in,
                              void* d_out, int out_size, void* d_ws, size_t ws_size,
                              hipStream_t stream) {
  const float* x     = (const float*)d_in[0];
  const float* graph = (const float*)d_in[1];
  const float* qWih  = (const float*)d_in[2];
  const float* qWhh  = (const float*)d_in[3];
  const float* qbih  = (const float*)d_in[4];
  const float* qbhh  = (const float*)d_in[5];
  const float* kWih  = (const float*)d_in[6];
  const float* kWhh  = (const float*)d_in[7];
  const float* kbih  = (const float*)d_in[8];
  const float* kbhh  = (const float*)d_in[9];
  const float* vWih  = (const float*)d_in[10];
  const float* vWhh  = (const float*)d_in[11];
  const float* vbih  = (const float*)d_in[12];
  const float* vbhh  = (const float*)d_in[13];
  float* out = (float*)d_out;

  // workspace layout
  const size_t off_qk   = 0;
  const size_t sz_qk    = (size_t)2 * 8 * QKROWS * DM * sizeof(unsigned short);   // 42.7 MB
  const size_t off_vT   = off_qk + sz_qk;
  const size_t sz_vT    = (size_t)8 * BATCH * TS * SP * sizeof(unsigned short);   // 34.6 MB
  const size_t off_attn = off_vT + sz_vT;
  unsigned short* qkbuf = (unsigned short*)((char*)d_ws + off_qk);
  unsigned short* vTbuf = (unsigned short*)((char*)d_ws + off_vT);
  unsigned short* atbuf = (unsigned short*)((char*)d_ws + off_attn);

  const int shm_k1 = 2 * 128 * HSTR * 2 + 128 * 33 * 4;   // 86528 B
  const int shm_k3 = 17408 + 8704 + 64 * 360 * 4;         // 118272 B
  const int shm_k4 = 46080 + 192 * 40 * 2;                // 61440 B

  (void)hipFuncSetAttribute((const void*)lstm_qk,
      hipFuncAttributeMaxDynamicSharedMemorySize, shm_k1);
  (void)hipFuncSetAttribute((const void*)attn_scores,
      hipFuncAttributeMaxDynamicSharedMemorySize, shm_k3);
  (void)hipFuncSetAttribute((const void*)attn_out,
      hipFuncAttributeMaxDynamicSharedMemorySize, shm_k4);

  // zero vT (padding cols must be 0 for K4)
  hipMemsetAsync((char*)d_ws + off_vT, 0, sz_vT, stream);

  lstm_qk<<<dim3(82, 16), 256, shm_k1, stream>>>(
      x, qWih, qWhh, qbih, qbhh, kWih, kWhh, kbih, kbhh, qkbuf);

  lstm_v<<<dim3((8 * NSEQ + 255) / 256), 256, 0, stream>>>(
      x, vWih, vWhh, vbih, vbhh, vTbuf);

  attn_scores<<<dim3(6, 256), 256, shm_k3, stream>>>(qkbuf, graph, atbuf);

  attn_out<<<dim3(6, 256), 256, shm_k4, stream>>>(atbuf, vTbuf, out);
}
